// GCN_30863634989345
// MI455X (gfx1250) — matrix operand order repaired and checked by reading, not compiled, not run on hardware
//
#include <hip/hip_runtime.h>
#include <stdint.h>

// ---------------------------------------------------------------------------
// GCN fused kernel for MI455X (gfx1250, wave32, WMMA bf16 16x16x32)
//   B*C = 32, N = 2048, F = H = O = 64
// ---------------------------------------------------------------------------

typedef __bf16 v16bf __attribute__((ext_vector_type(16)));
typedef float  v8f   __attribute__((ext_vector_type(8)));
typedef float  f32x4 __attribute__((ext_vector_type(4)));
typedef unsigned int u32x4 __attribute__((ext_vector_type(4)));

#define BC   32
#define NSEQ 2048
#define FDIM 64
#define SWAPX16 0x401F   // ds_swizzle: xor=0x10, or=0, and=0x1f -> swap lane halves

struct Frag { uint32_t u[8]; };   // one 16x32 (A) or 32x16 (B) bf16 operand

__device__ __forceinline__ v16bf frag_bf(const Frag& f) {
  union { Frag f; v16bf v; } c; c.f = f; return c.v;
}

// f32 pair -> packed bf16 dword (RNE) in ONE VALU op (gfx1250 hw cvt).
__device__ __forceinline__ uint32_t pk_bf16(float a, float b) {
  uint32_t r;
  asm("v_cvt_pk_bf16_f32 %0, %1, %2" : "=v"(r) : "v"(a), "v"(b));
  return r;
}

// ReLU on a packed bf16 pair: one VOP3P op (max with +0.0 in both halves).
__device__ __forceinline__ uint32_t relu_pk_bf16(uint32_t v) {
  uint32_t r;
  asm("v_pk_max_num_bf16 %0, %1, 0" : "=v"(r) : "v"(v));
  return r;
}

__device__ __forceinline__ v8f wmma_bf16(const Frag& a, const Frag& b, v8f c) {
  return __builtin_amdgcn_wmma_f32_16x16x32_bf16(
      false, frag_bf(a), false, frag_bf(b), (short)0, c, false, false);
}

// A operand (16x32, MxK) from row-major bf16 [row][k], stride in elements.
// lane<16: row0+r, k {k0..k0+15}; lane>=16: k {k0+16..k0+31}
__device__ __forceinline__ Frag loadA(const uint16_t* base, int stride,
                                      int row0, int k0, int lane) {
  const bool lo = lane < 16;
  const int  r  = lane & 15;
  const uint16_t* p = base + (long)(row0 + r) * stride + k0 + (lo ? 0 : 16);
  u32x4 q0 = *(const u32x4*)p;
  u32x4 q1 = *(const u32x4*)(p + 8);
  Frag f;
#pragma unroll
  for (int t = 0; t < 4; ++t) { f.u[t] = q0[t]; f.u[t + 4] = q1[t]; }
  return f;
}

// B operand (32x16, KxN): B[k][j] = src[row0+j][k]   (src row-major bf16)
// lane<16: row row0+j, k {k0..k0+15}; lane>=16: row row0+j-16, k {k0+16..k0+31}
__device__ __forceinline__ Frag loadB(const uint16_t* base, int stride,
                                      int row0, int k0, int lane) {
  const bool lo = lane < 16;
  const int  r  = lane & 15;
  const uint16_t* p = base + (long)(row0 + r) * stride + k0 + (lo ? 0 : 16);
  u32x4 q0 = *(const u32x4*)p;
  u32x4 q1 = *(const u32x4*)(p + 8);
  Frag f;
#pragma unroll
  for (int t = 0; t < 4; ++t) { f.u[t] = q0[t]; f.u[t + 4] = q1[t]; }
  return f;
}

// A operand built from row-major fp32 source (on-the-fly hw bf16 convert)
__device__ __forceinline__ Frag loadA_f32(const float* base, int stride,
                                          int row0, int k0, int lane) {
  const bool lo = lane < 16;
  const int  r  = lane & 15;
  const float* p = base + (long)(row0 + r) * stride + k0 + (lo ? 0 : 16);
  f32x4 a0 = *(const f32x4*)p;
  f32x4 a1 = *(const f32x4*)(p + 4);
  f32x4 a2 = *(const f32x4*)(p + 8);
  f32x4 a3 = *(const f32x4*)(p + 12);
  Frag f;
  f.u[0] = pk_bf16(a0[0], a0[1]); f.u[1] = pk_bf16(a0[2], a0[3]);
  f.u[2] = pk_bf16(a1[0], a1[1]); f.u[3] = pk_bf16(a1[2], a1[3]);
  f.u[4] = pk_bf16(a2[0], a2[1]); f.u[5] = pk_bf16(a2[2], a2[3]);
  f.u[6] = pk_bf16(a3[0], a3[1]); f.u[7] = pk_bf16(a3[2], a3[3]);
  return f;
}

// B operand from row-major fp32 source
__device__ __forceinline__ Frag loadB_f32(const float* base, int stride,
                                          int row0, int k0, int lane) {
  const bool lo = lane < 16;
  const int  r  = lane & 15;
  const float* p = base + (long)(row0 + r) * stride + k0 + (lo ? 0 : 16);
  f32x4 a0 = *(const f32x4*)p;
  f32x4 a1 = *(const f32x4*)(p + 4);
  f32x4 a2 = *(const f32x4*)(p + 8);
  f32x4 a3 = *(const f32x4*)(p + 12);
  Frag f;
  f.u[0] = pk_bf16(a0[0], a0[1]); f.u[1] = pk_bf16(a0[2], a0[3]);
  f.u[2] = pk_bf16(a1[0], a1[1]); f.u[3] = pk_bf16(a1[2], a1[3]);
  f.u[4] = pk_bf16(a2[0], a2[1]); f.u[5] = pk_bf16(a2[2], a2[3]);
  f.u[6] = pk_bf16(a3[0], a3[1]); f.u[7] = pk_bf16(a3[2], a3[3]);
  return f;
}

// Pack a C/D f32 tile (8 regs, row pairs) into 4 bf16-pair dwords
__device__ __forceinline__ void pack_tile(const v8f& s, uint32_t pk[4]) {
  pk[0] = pk_bf16(s[0], s[1]);
  pk[1] = pk_bf16(s[2], s[3]);
  pk[2] = pk_bf16(s[4], s[5]);
  pk[3] = pk_bf16(s[6], s[7]);
}

// Two transposed C/D tiles (Kx16 halves) -> one 32x16 B operand, in-register.
// Uses a half-wave exchange: lanes 0..15 <-> 16..31 (ds_swizzle SWAPX16).
__device__ __forceinline__ Frag combine_halves(const uint32_t p0[4],
                                               const uint32_t p1[4], bool lo) {
  Frag b;
#pragma unroll
  for (int t = 0; t < 4; ++t) {
    uint32_t s0 = (uint32_t)__builtin_amdgcn_ds_swizzle((int)p0[t], SWAPX16);
    uint32_t s1 = (uint32_t)__builtin_amdgcn_ds_swizzle((int)p1[t], SWAPX16);
    b.u[t]     = lo ? p0[t] : s1;   // K 0..7  | K 16..23
    b.u[t + 4] = lo ? s0 : p1[t];   // K 8..15 | K 24..31
  }
  return b;
}

// ---------------------------------------------------------------------------
// Kernel 0: x -> xT (bf16, [bc][f][n]) and W3 -> bf16
// ---------------------------------------------------------------------------
#define XT_TOTAL (BC * FDIM * NSEQ)   // 4,194,304

__global__ __launch_bounds__(256) void gcn_prep(const float* __restrict__ x,
                                                const float* __restrict__ W3,
                                                uint16_t* __restrict__ xT,
                                                uint16_t* __restrict__ w3b) {
  int idx = blockIdx.x * 256 + threadIdx.x;
  if (idx < XT_TOTAL) {
    int n  = idx & (NSEQ - 1);
    int f  = (idx >> 11) & (FDIM - 1);
    int bc = idx >> 17;
    float v = x[((long)bc * NSEQ + n) * FDIM + f];
    xT[((long)bc * FDIM + f) * NSEQ + n] = (uint16_t)pk_bf16(v, 0.f);
  } else if (idx < XT_TOTAL + FDIM * FDIM) {
    int j = idx - XT_TOTAL;
    w3b[j] = (uint16_t)pk_bf16(W3[j], 0.f);
  }
}

// ---------------------------------------------------------------------------
// Kernel 1: e1 = x@W1^T + b1, e2 = x@W2^T + b2  (WMMA, bf16 out, row-major)
// One wave per 16 rows; 4 waves per block.
// ---------------------------------------------------------------------------
__global__ __launch_bounds__(128) void gcn_embed(const float* __restrict__ x,
                                                 const float* __restrict__ W1,
                                                 const float* __restrict__ b1,
                                                 const float* __restrict__ W2,
                                                 const float* __restrict__ b2,
                                                 uint16_t* __restrict__ e1,
                                                 uint16_t* __restrict__ e2) {
  const int lane = threadIdx.x & 31;
  const int wave = threadIdx.x >> 5;
  const int bc   = blockIdx.x >> 5;          // 32 bc
  const int grp  = blockIdx.x & 31;          // 32 row-groups of 64
  const int n0   = grp * 64 + wave * 16;
  const bool lo  = lane < 16;

  const float* xbc = x + (long)bc * NSEQ * FDIM;
  Frag ax0 = loadA_f32(xbc, FDIM, n0, 0, lane);
  Frag ax1 = loadA_f32(xbc, FDIM, n0, 32, lane);

  const float*    Wp[2] = { W1, W2 };
  const float*    bp[2] = { b1, b2 };
  uint16_t*       ep[2] = { e1, e2 };

#pragma unroll
  for (int wsel = 0; wsel < 2; ++wsel) {
#pragma unroll
    for (int hb = 0; hb < 4; ++hb) {
      // B[k=f][j=h] = W[hb*16+j][f]
      Frag bw0 = loadB_f32(Wp[wsel], FDIM, hb * 16, 0, lane);
      Frag bw1 = loadB_f32(Wp[wsel], FDIM, hb * 16, 32, lane);
      v8f acc = {};
      acc = wmma_bf16(ax0, bw0, acc);
      acc = wmma_bf16(ax1, bw1, acc);
      float bias = bp[wsel][hb * 16 + (lane & 15)];   // column = lane
      // store: column h fixed per lane, 8 rows n (pairs packed by hw cvt)
      uint16_t* dst = ep[wsel] +
          ((long)bc * NSEQ + n0 + (lo ? 0 : 8)) * FDIM + hb * 16 + (lane & 15);
#pragma unroll
      for (int r = 0; r < 8; r += 2) {
        uint32_t pr = pk_bf16(acc[r] + bias, acc[r + 1] + bias);
        dst[(long)r * FDIM]       = (uint16_t)pr;
        dst[(long)(r + 1) * FDIM] = (uint16_t)(pr >> 16);
      }
    }
  }
}

// ---------------------------------------------------------------------------
// Kernel 2: fused  h = relu(e1@e2^T) @ x ;  out = h@W3^T + b3
// Per wave: 16 output rows (n). m-loop over 2048 in steps of 32.
// Scores computed transposed (S^T = E2 * E1^T) so the relu'd tile can be fed
// straight back as the next WMMA's B operand via a half-wave swizzle.
// ---------------------------------------------------------------------------
__global__ __launch_bounds__(128) void gcn_main(const uint16_t* __restrict__ e1,
                                                const uint16_t* __restrict__ e2,
                                                const uint16_t* __restrict__ xT,
                                                const uint16_t* __restrict__ w3b,
                                                const float* __restrict__ b3,
                                                float* __restrict__ out) {
  const int lane = threadIdx.x & 31;
  const int wave = threadIdx.x >> 5;
  const int bc   = blockIdx.x >> 5;
  const int grp  = blockIdx.x & 31;
  const int n0   = grp * 64 + wave * 16;
  const bool lo  = lane < 16;

  const uint16_t* e1bc = e1 + (long)bc * NSEQ * FDIM;
  const uint16_t* e2bc = e2 + (long)bc * NSEQ * FDIM;
  const uint16_t* xTbc = xT + (long)bc * FDIM * NSEQ;

  // B operand of the score WMMA: B[k=h][j=n] = e1[n0+j][h]; reused all loop.
  Frag be1_0 = loadB(e1bc, FDIM, n0, 0, lane);
  Frag be1_1 = loadB(e1bc, FDIM, n0, 32, lane);

  v8f hacc0 = {}, hacc1 = {}, hacc2 = {}, hacc3 = {};   // H^T[f][n], f blocks

#pragma unroll 1
  for (int m0 = 0; m0 < NSEQ; m0 += 32) {
    // ---- scores: S^T[m][n] = sum_h e2[m][h] * e1[n][h]  (two 16-row tiles)
    Frag a00 = loadA(e2bc, FDIM, m0,      0,  lane);
    Frag a01 = loadA(e2bc, FDIM, m0,      32, lane);
    Frag a10 = loadA(e2bc, FDIM, m0 + 16, 0,  lane);
    Frag a11 = loadA(e2bc, FDIM, m0 + 16, 32, lane);
    v8f s0 = {}, s1 = {};
    s0 = wmma_bf16(a00, be1_0, s0);
    s0 = wmma_bf16(a01, be1_1, s0);
    s1 = wmma_bf16(a10, be1_0, s1);
    s1 = wmma_bf16(a11, be1_1, s1);
    // ---- pack to bf16 (hw cvt), then ReLU on packed pairs (v_pk_max)
    uint32_t pk0[4], pk1[4];
    pack_tile(s0, pk0);
    pack_tile(s1, pk1);
#pragma unroll
    for (int t = 0; t < 4; ++t) {
      pk0[t] = relu_pk_bf16(pk0[t]);
      pk1[t] = relu_pk_bf16(pk1[t]);
    }
    Frag bp = combine_halves(pk0, pk1, lo);
    // ---- H^T[f][n] += x^T[f][m] * P^T[m][n]
    Frag ax0 = loadA(xTbc, NSEQ, 0,  m0, lane);
    Frag ax1 = loadA(xTbc, NSEQ, 16, m0, lane);
    Frag ax2 = loadA(xTbc, NSEQ, 32, m0, lane);
    Frag ax3 = loadA(xTbc, NSEQ, 48, m0, lane);
    hacc0 = wmma_bf16(ax0, bp, hacc0);
    hacc1 = wmma_bf16(ax1, bp, hacc1);
    hacc2 = wmma_bf16(ax2, bp, hacc2);
    hacc3 = wmma_bf16(ax3, bp, hacc3);
  }

  // ---- projection: out^T[o][n] = sum_f W3[o][f] * H^T[f][n] + b3[o]
  uint32_t ph0[4], ph1[4], ph2[4], ph3[4];
  pack_tile(hacc0, ph0);
  pack_tile(hacc1, ph1);
  pack_tile(hacc2, ph2);
  pack_tile(hacc3, ph3);
  Frag bh0 = combine_halves(ph0, ph1, lo);   // K = f 0..31
  Frag bh1 = combine_halves(ph2, ph3, lo);   // K = f 32..63

#pragma unroll
  for (int ob = 0; ob < 4; ++ob) {
    Frag aw0 = loadA(w3b, FDIM, ob * 16, 0, lane);
    Frag aw1 = loadA(w3b, FDIM, ob * 16, 32, lane);
    v8f oa = {};
    oa = wmma_bf16(aw0, bh0, oa);
    oa = wmma_bf16(aw1, bh1, oa);
    // store: lane j holds column n = n0+(lane&15); rows r are o = ob0+r
    const int n   = n0 + (lane & 15);
    const int ob0 = ob * 16 + (lo ? 0 : 8);
    float* op = out + ((long)bc * NSEQ + n) * FDIM + ob0;
    f32x4 v0, v1;
#pragma unroll
    for (int r = 0; r < 4; ++r) v0[r] = oa[r]     + b3[ob0 + r];
#pragma unroll
    for (int r = 0; r < 4; ++r) v1[r] = oa[r + 4] + b3[ob0 + 4 + r];
    *(f32x4*)op       = v0;
    *(f32x4*)(op + 4) = v1;
  }
}

// ---------------------------------------------------------------------------
extern "C" void kernel_launch(void* const* d_in, const int* in_sizes, int n_in,
                              void* d_out, int out_size, void* d_ws, size_t ws_size,
                              hipStream_t stream) {
  const float* x  = (const float*)d_in[0];
  const float* W1 = (const float*)d_in[1];
  const float* b1 = (const float*)d_in[2];
  const float* W2 = (const float*)d_in[3];
  const float* b2 = (const float*)d_in[4];
  const float* W3 = (const float*)d_in[5];
  const float* b3 = (const float*)d_in[6];
  float* out = (float*)d_out;

  const long EN = (long)BC * NSEQ * FDIM;       // 4,194,304 elements
  uint16_t* e1  = (uint16_t*)d_ws;
  uint16_t* e2  = e1 + EN;
  uint16_t* xT  = e2 + EN;
  uint16_t* w3b = xT + EN;

  gcn_prep <<<(XT_TOTAL + FDIM * FDIM) / 256, 256, 0, stream>>>(x, W3, xT, w3b);
  gcn_embed<<<BC * (NSEQ / 64), 128, 0, stream>>>(x, W1, b1, W2, b2, e1, e2);
  gcn_main <<<BC * (NSEQ / 64), 128, 0, stream>>>(e1, e2, xT, w3b, b3, out);
}
